// RelPositionMultiHeadAttention_85212151152795
// MI455X (gfx1250) — matrix operand order repaired and checked by reading, not compiled, not run on hardware
//
#include <hip/hip_runtime.h>

// Problem constants (from reference)
#define TT 1024
#define FF 512
#define HH 8
#define DD 64
#define PPOS 2047
#define BB 8

typedef __attribute__((ext_vector_type(16))) __bf16 v16bf;
typedef __attribute__((ext_vector_type(8)))  float  v8f;

// D = A(16x32 bf16) x B(32x16 bf16) + C(16x16 f32)
__device__ __forceinline__ v8f wmma_bf16(const v16bf& a, const v16bf& b, const v8f& c) {
  return __builtin_amdgcn_wmma_f32_16x16x32_bf16(false, a, false, b, (short)0, c, false, false);
}

// A fragment (16x32, MxK): lane m=lane&15, kb = lane<16 ? 0 : 8
// elements: k = kb..kb+7, then kb+16..kb+23  (two contiguous 16B runs -> ds_load_b128)
__device__ __forceinline__ void frag_a(const __bf16* base, int stride, v16bf& a) {
  int lane = threadIdx.x & 31;
  const __bf16* p = base + (lane & 15) * stride + ((lane < 16) ? 0 : 8);
#pragma unroll
  for (int i = 0; i < 8; ++i) a[i] = p[i];
#pragma unroll
  for (int i = 0; i < 8; ++i) a[8 + i] = p[16 + i];
}

// B fragment (32x16, KxN) stored in LDS as [n][k] rows: lane n=lane&15,
// kb = lane<16 ? 0 : 8, elements k = kb..kb+7, then kb+16..kb+23
__device__ __forceinline__ void frag_b(const __bf16* base, int stride, v16bf& b) {
  int lane = threadIdx.x & 31;
  const __bf16* p = base + (lane & 15) * stride + ((lane < 16) ? 0 : 8);
#pragma unroll
  for (int i = 0; i < 8; ++i) b[i] = p[i];
#pragma unroll
  for (int i = 0; i < 8; ++i) b[8 + i] = p[16 + i];
}

// ---------------------------------------------------------------------------
// C[M,512] = A[M,512] @ W[512,512]^T (+ bias).  ABF: A is bf16 else fp32.
// OMODE 0: bf16 out. 1: fp32 out. 2: dual bf16 out (+pos_bias_u / +pos_bias_v).
// Block = 256 threads (8 waves), tile 128(M) x 64(N), BK=32.
// ---------------------------------------------------------------------------
template <bool ABF, int OMODE>
__global__ __launch_bounds__(256) void proj_gemm(
    const void* __restrict__ Ain, const float* __restrict__ W,
    const float* __restrict__ bias, void* __restrict__ out0,
    void* __restrict__ out1, const float* __restrict__ ub,
    const float* __restrict__ vb, int M) {
  const int K = FF, N = FF;
  __shared__ __bf16 As[128 * 32];
  __shared__ __bf16 Bs[64 * 32];

  const int t = threadIdx.x;
  const int lane = t & 31, wave = t >> 5;
  const int m0 = blockIdx.y * 128, n0 = blockIdx.x * 64;
  const int wm = (wave >> 1) * 32;  // 4 waves along M
  const int wn = (wave & 1) * 32;   // 2 waves along N

  v8f acc[2][2] = {};

  for (int k0 = 0; k0 < K; k0 += 32) {
    // A tile: 128x32 -> bf16 LDS (1024 x float4-equivalents, 4 per thread)
#pragma unroll
    for (int i = 0; i < 4; ++i) {
      int q = i * 256 + t;
      int row = q >> 3, c4 = (q & 7) * 4;
      int gm = m0 + row; if (gm > M - 1) gm = M - 1;   // clamp (M=2047 case)
      size_t off = (size_t)gm * K + k0 + c4;
      if constexpr (ABF) {
        const __bf16* p = (const __bf16*)Ain + off;
#pragma unroll
        for (int j = 0; j < 4; ++j) As[row * 32 + c4 + j] = p[j];
      } else {
        const float4 v = *(const float4*)((const float*)Ain + off);
        As[row * 32 + c4 + 0] = (__bf16)v.x;
        As[row * 32 + c4 + 1] = (__bf16)v.y;
        As[row * 32 + c4 + 2] = (__bf16)v.z;
        As[row * 32 + c4 + 3] = (__bf16)v.w;
      }
    }
    // W tile: 64x32 fp32 -> bf16 LDS (512 float4, 2 per thread)
#pragma unroll
    for (int i = 0; i < 2; ++i) {
      int q = i * 256 + t;
      int row = q >> 3, c4 = (q & 7) * 4;
      const float4 v = *(const float4*)(W + (size_t)(n0 + row) * K + k0 + c4);
      Bs[row * 32 + c4 + 0] = (__bf16)v.x;
      Bs[row * 32 + c4 + 1] = (__bf16)v.y;
      Bs[row * 32 + c4 + 2] = (__bf16)v.z;
      Bs[row * 32 + c4 + 3] = (__bf16)v.w;
    }
    if (k0 + 32 < K)  // CDNA5 global_prefetch_b8 for next W tile
      __builtin_prefetch((const char*)(W + (size_t)(n0 + (t >> 3)) * K + k0 + 32), 0, 3);
    __syncthreads();

    v16bf a0, a1, b0, b1;
    frag_a(As + wm * 32, 32, a0);
    frag_a(As + (wm + 16) * 32, 32, a1);
    frag_b(Bs + wn * 32, 32, b0);
    frag_b(Bs + (wn + 16) * 32, 32, b1);
    acc[0][0] = wmma_bf16(a0, b0, acc[0][0]);
    acc[0][1] = wmma_bf16(a0, b1, acc[0][1]);
    acc[1][0] = wmma_bf16(a1, b0, acc[1][0]);
    acc[1][1] = wmma_bf16(a1, b1, acc[1][1]);
    __syncthreads();
  }

  // Epilogue: C layout — lane n = lane&15, VGPR g holds row g + 8*(lane>>4)
  const int lh = lane >> 4, lc = lane & 15;
#pragma unroll
  for (int tm = 0; tm < 2; ++tm)
#pragma unroll
    for (int tn = 0; tn < 2; ++tn)
#pragma unroll
      for (int g = 0; g < 8; ++g) {
        int m = m0 + wm + tm * 16 + g + 8 * lh;
        int n = n0 + wn + tn * 16 + lc;
        if (m < M) {
          float v = acc[tm][tn][g] + (bias ? bias[n] : 0.f);
          if constexpr (OMODE == 1) {
            ((float*)out0)[(size_t)m * N + n] = v;
          } else if constexpr (OMODE == 0) {
            ((__bf16*)out0)[(size_t)m * N + n] = (__bf16)v;
          } else {
            ((__bf16*)out0)[(size_t)m * N + n] = (__bf16)(v + ub[n]);
            ((__bf16*)out1)[(size_t)m * N + n] = (__bf16)(v + vb[n]);
          }
        }
      }
}

// ---------------------------------------------------------------------------
// Fused rel-pos flash attention. One wave per (b, h, 16-row q tile).
// rel_shift identity: bd[i,j] = sum_d Qv[i,d] * Pp[(j-i)+(T-1), d]
// Per 32-key chunk the needed Pp rows form a 47-wide window ->
// compute M = Qv_tile @ Pwin^T via WMMA, then gather bd[i,j] = M[i, 15+j-i].
// ---------------------------------------------------------------------------
__global__ __launch_bounds__(32) void flash_attn(
    const __bf16* __restrict__ Qu, const __bf16* __restrict__ Qv,
    const __bf16* __restrict__ Kb, const __bf16* __restrict__ Vb,
    const __bf16* __restrict__ Pp, __bf16* __restrict__ AO) {
  const float scale = 0.125f;  // 1/sqrt(64)
  const int lane = threadIdx.x;
  const int i0 = blockIdx.x * 16, h = blockIdx.y, b = blockIdx.z;

  __shared__ __bf16 sQ[16 * 64];
  __shared__ __bf16 sK[32 * 64];
  __shared__ __bf16 sVT[64 * 32];   // V transposed: [d][j]
  __shared__ __bf16 sP[48 * 64];    // position window
  __shared__ float  sS[16 * 32];
  __shared__ float  sBD[16 * 48];
  __shared__ __bf16 sPr[16 * 32];   // softmax probs (A-operand staging)
  __shared__ float  sMx[16], sSum[16], sCorr[16];

  // Stage Qu tile -> register A fragments (held whole kernel)
  {
    int r = lane >> 1, half = lane & 1;
    const __bf16* src = Qu + ((size_t)(b * TT + i0 + r) * FF + h * DD + half * 32);
#pragma unroll
    for (int i = 0; i < 32; ++i) sQ[r * 64 + half * 32 + i] = src[i];
  }
  __syncthreads();
  v16bf aU0, aU1;
  frag_a(sQ, 64, aU0);
  frag_a(sQ + 32, 64, aU1);
  __syncthreads();
  {
    int r = lane >> 1, half = lane & 1;
    const __bf16* src = Qv + ((size_t)(b * TT + i0 + r) * FF + h * DD + half * 32);
#pragma unroll
    for (int i = 0; i < 32; ++i) sQ[r * 64 + half * 32 + i] = src[i];
  }
  __syncthreads();
  v16bf aV0, aV1;
  frag_a(sQ, 64, aV0);
  frag_a(sQ + 32, 64, aV1);

  if (lane < 16) { sMx[lane] = -3.0e38f; sSum[lane] = 0.f; }
  __syncthreads();

  v8f o[4] = {};

  for (int c = 0; c < TT / 32; ++c) {
    const int j0 = c * 32;
    // K tile [32][64]
    {
      const __bf16* src = Kb + ((size_t)(b * TT + j0 + lane) * FF + h * DD);
#pragma unroll
      for (int i = 0; i < 64; ++i) sK[lane * 64 + i] = src[i];
    }
    // V tile transposed [64][32]
    {
      const __bf16* src = Vb + ((size_t)(b * TT + j0 + lane) * FF + h * DD);
#pragma unroll
      for (int d = 0; d < 64; ++d) sVT[d * 32 + lane] = src[d];
    }
    // P window: rows rel = (T-1)+j0-i0-15 .. +32  (clamped; pads unused)
    {
      int base = (TT - 1) + j0 - i0 - 15;
#pragma unroll
      for (int w = lane; w < 48; w += 32) {
        int rel = base + w;
        rel = rel < 0 ? 0 : (rel > PPOS - 1 ? PPOS - 1 : rel);
        const __bf16* src = Pp + ((size_t)rel * FF + h * DD);
#pragma unroll
        for (int i = 0; i < 64; ++i) sP[w * 64 + i] = src[i];
      }
    }
    __syncthreads();

    const int lh = lane >> 4, lc = lane & 15;

    // BD raw = Qv_tile @ Pwin^T  -> [16,48]
#pragma unroll
    for (int nt = 0; nt < 3; ++nt) {
      v16bf b0, b1;
      frag_b(sP + nt * 16 * 64, 64, b0);
      frag_b(sP + nt * 16 * 64 + 32, 64, b1);
      v8f m = {};
      m = wmma_bf16(aV0, b0, m);
      m = wmma_bf16(aV1, b1, m);
#pragma unroll
      for (int g = 0; g < 8; ++g) sBD[(g + 8 * lh) * 48 + nt * 16 + lc] = m[g];
    }
    // S = Qu_tile @ K_tile^T -> [16,32]
#pragma unroll
    for (int nt = 0; nt < 2; ++nt) {
      v16bf b0, b1;
      frag_b(sK + nt * 16 * 64, 64, b0);
      frag_b(sK + nt * 16 * 64 + 32, 64, b1);
      v8f s = {};
      s = wmma_bf16(aU0, b0, s);
      s = wmma_bf16(aU1, b1, s);
#pragma unroll
      for (int g = 0; g < 8; ++g) sS[(g + 8 * lh) * 32 + nt * 16 + lc] = s[g];
    }
    __syncthreads();

    // Online softmax: lanes 0..15 own one row each; diagonal gather of BD here
    if (lane < 16) {
      const int r = lane;
      float mx = sMx[r], nmx = mx;
      float sv[32];
#pragma unroll
      for (int j = 0; j < 32; ++j) {
        float s = scale * (sS[r * 32 + j] + sBD[r * 48 + 15 + j - r]);
        sv[j] = s;
        nmx = fmaxf(nmx, s);
      }
      float corr = __expf(mx - nmx);
      float sum = sSum[r] * corr;
#pragma unroll
      for (int j = 0; j < 32; ++j) {
        float p = __expf(sv[j] - nmx);
        sum += p;
        sPr[r * 32 + j] = (__bf16)p;
      }
      sMx[r] = nmx; sSum[r] = sum; sCorr[r] = corr;
    }
    __syncthreads();

    // Rescale O and accumulate O += P @ V
    v16bf aP;
    frag_a(sPr, 32, aP);
#pragma unroll
    for (int g = 0; g < 8; ++g) {
      float cc = sCorr[g + 8 * lh];
#pragma unroll
      for (int dt = 0; dt < 4; ++dt) o[dt][g] *= cc;
    }
#pragma unroll
    for (int dt = 0; dt < 4; ++dt) {
      v16bf bv;
      frag_b(sVT + dt * 16 * 32, 32, bv);
      o[dt] = wmma_bf16(aP, bv, o[dt]);
    }
    __syncthreads();
  }

  // Normalize and write AO[b, t, h*D + d] (bf16)
  const int lh = lane >> 4, lc = lane & 15;
#pragma unroll
  for (int g = 0; g < 8; ++g) {
    int row = g + 8 * lh;
    float inv = 1.f / sSum[row];
#pragma unroll
    for (int dt = 0; dt < 4; ++dt)
      AO[(size_t)(b * TT + i0 + row) * FF + h * DD + dt * 16 + lc] =
          (__bf16)(o[dt][g] * inv);
  }
}

// ---------------------------------------------------------------------------
extern "C" void kernel_launch(void* const* d_in, const int* in_sizes, int n_in,
                              void* d_out, int out_size, void* d_ws,
                              size_t ws_size, hipStream_t stream) {
  const float* query   = (const float*)d_in[0];
  const float* key     = (const float*)d_in[1];
  const float* value   = (const float*)d_in[2];
  const float* pos_emb = (const float*)d_in[3];
  const float* Wq = (const float*)d_in[4];
  const float* bq = (const float*)d_in[5];
  const float* Wk = (const float*)d_in[6];
  const float* bk = (const float*)d_in[7];
  const float* Wv = (const float*)d_in[8];
  const float* bv = (const float*)d_in[9];
  const float* Wp = (const float*)d_in[10];
  const float* Wo = (const float*)d_in[11];
  const float* bo = (const float*)d_in[12];
  const float* pbu = (const float*)d_in[13];
  const float* pbv = (const float*)d_in[14];

  const size_t MQ = (size_t)BB * TT;  // 8192
  char* ws = (char*)d_ws;
  __bf16* Qu = (__bf16*)ws; ws += MQ * FF * 2;
  __bf16* Qv = (__bf16*)ws; ws += MQ * FF * 2;
  __bf16* Kb = (__bf16*)ws; ws += MQ * FF * 2;
  __bf16* Vb = (__bf16*)ws; ws += MQ * FF * 2;
  __bf16* AO = (__bf16*)ws; ws += MQ * FF * 2;
  __bf16* Pb = (__bf16*)ws; ws += (size_t)2048 * FF * 2;

  dim3 blk(256);
  dim3 gq(FF / 64, (unsigned)((MQ + 127) / 128));
  dim3 gp(FF / 64, (PPOS + 127) / 128);

  // Projections (fp32 in -> bf16 out), Q gets both bias-u and bias-v variants
  proj_gemm<false, 2><<<gq, blk, 0, stream>>>(query, Wq, bq, Qu, Qv, pbu, pbv, (int)MQ);
  proj_gemm<false, 0><<<gq, blk, 0, stream>>>(key, Wk, bk, Kb, nullptr, nullptr, nullptr, (int)MQ);
  proj_gemm<false, 0><<<gq, blk, 0, stream>>>(value, Wv, bv, Vb, nullptr, nullptr, nullptr, (int)MQ);
  proj_gemm<false, 0><<<gp, blk, 0, stream>>>(pos_emb, Wp, nullptr, Pb, nullptr, nullptr, nullptr, PPOS);

  // Fused rel-pos flash attention
  flash_attn<<<dim3(TT / 16, HH, BB), dim3(32), 0, stream>>>(Qu, Qv, Kb, Vb, Pb, AO);

  // Output projection (bf16 in -> fp32 out)
  proj_gemm<true, 1><<<gq, blk, 0, stream>>>(AO, Wo, bo, d_out, nullptr, nullptr, nullptr, (int)MQ);
}
